// WavenetModel_77257871721126
// MI455X (gfx1250) — matrix operand order repaired and checked by reading, not compiled, not run on hardware
//
#include <hip/hip_runtime.h>
#include <hip/hip_bf16.h>
#include <math.h>

typedef __attribute__((ext_vector_type(16))) __bf16 v16bf;
typedef __attribute__((ext_vector_type(8)))  float  v8f;

constexpr int Tlen = 16384;
constexpr int Bsz  = 8;
constexpr int R    = Bsz * Tlen;   // 131072 flattened (b,t) rows
constexpr int RC   = 64;           // residual channels
constexpr int OCH  = 256;

// bf16 weight workspace layout (elements)
constexpr int W_IL  = 0;                    // [64][512]   k = tap*256+ic
constexpr int W_G   = W_IL + 64 * 512;      // [20][128][128] k = tap*64+ic
constexpr int W_S   = W_G + 20 * 128 * 128; // [20][64][64]
constexpr int W_O1  = W_S + 20 * 64 * 64;   // [256][64]
constexpr int W_O2  = W_O1 + 256 * 64;      // [256][256]
constexpr int W_TOT = W_O2 + 256 * 256;     // 524288 elements

// ---- CDNA5 async global->LDS copy (16B per lane), ASYNCcnt tracked ----
__device__ inline void async_ld16(const void* gsrc, const void* ldst) {
  unsigned l = (unsigned)(unsigned long long)ldst;       // addr[31:0] = LDS offset
  unsigned long long g = (unsigned long long)gsrc;
  asm volatile("global_load_async_to_lds_b128 %0, %1, off"
               :: "v"(l), "v"(g) : "memory");
}
__device__ inline void async_wait0() {
  asm volatile("s_wait_asynccnt 0x0" ::: "memory");
}

// ---- WMMA fragment gathers (contiguous k = 16*hi + e in A and B, wave32) ----
__device__ inline v16bf frag_a(const __bf16* A, int lda, int m, int kbase, int hi) {
  const __bf16* p = A + m * lda + kbase;
  v16bf a;
#pragma unroll
  for (int e = 0; e < 16; ++e) a[e] = p[16 * hi + e];
  return a;
}
__device__ inline v16bf frag_b(const __bf16* Bt, int ldk, int n, int kbase, int hi) {
  const __bf16* p = Bt + n * ldk + kbase + 16 * hi;
  v16bf b;
#pragma unroll
  for (int e = 0; e < 16; ++e) b[e] = p[e];
  return b;
}
__device__ inline v8f wmma_bf16(v16bf a, v16bf b, v8f c) {
  return __builtin_amdgcn_wmma_f32_16x16x32_bf16(false, a, false, b, (short)0, c, false, false);
}
// C/D: value (VGPR r, lane) = D[M = r + 8*(lane>>4)][N = lane&15]

// ------------- prologue: convert + pre-transpose all weights to bf16 Bt[N][K] -------------
__global__ __launch_bounds__(256) void wn_cvt(const float* __restrict__ il_w,
                                              const float* __restrict__ rb_w1,
                                              const float* __restrict__ rb_w2,
                                              const float* __restrict__ ol_w1,
                                              const float* __restrict__ ol_w2,
                                              __bf16* __restrict__ wb) {
  int i = blockIdx.x * 256 + threadIdx.x;      // 524288 total
  if (i < 64 * 512) {                          // input conv
    int n = i >> 9, k = i & 511;
    wb[W_IL + i] = (__bf16)il_w[n * 512 + (k & 255) * 2 + (k >> 8)];
    return;
  }
  i -= 64 * 512;
  if (i < 20 * 128 * 128) {                    // gate convs
    int l = i >> 14, r2 = i & 16383;
    int n = r2 >> 7, k = r2 & 127;
    wb[W_G + i] = (__bf16)rb_w1[l * 16384 + n * 128 + (k & 63) * 2 + (k >> 6)];
    return;
  }
  i -= 20 * 128 * 128;
  if (i < 20 * 64 * 64) { wb[W_S + i] = (__bf16)rb_w2[i]; return; }   // already [n][k]
  i -= 20 * 64 * 64;
  if (i < 256 * 64) { wb[W_O1 + i] = (__bf16)ol_w1[i]; return; }
  i -= 256 * 64;
  wb[W_O2 + i] = (__bf16)ol_w2[i];
}

// ---------------- input layer: causal conv K=2, 256 -> 64 ----------------
__global__ __launch_bounds__(256) void wn_input(const float* __restrict__ x,
                                                const __bf16* __restrict__ wIL,  // [64][512] Bt
                                                const float* __restrict__ bias,
                                                __bf16* __restrict__ o) {
  __shared__ __bf16 smA[64 * 128];
  __shared__ __bf16 smB[64 * 128];
  const int tid = threadIdx.x, lane = tid & 31, hi = lane >> 4, wv = tid >> 5;
  const int rowBase = blockIdx.x * 64;

  v8f acc[2] = {};
  for (int c = 0; c < 4; ++c) {                // K chunks of 128
    const int tap = c >> 1;
    const int icbase = (c & 1) * 128;
    // weights: async copy, rows of 256B -> 16 x b128, 1024 transfers
    for (int s = 0; s < 4; ++s) {
      int tIdx = s * 256 + tid;
      int n = tIdx >> 4, cc = tIdx & 15;
      async_ld16(wIL + n * 512 + c * 128 + cc * 8, smB + n * 128 + cc * 8);
    }
    // activations: fp32 x -> bf16 (conversion path, can't async)
    for (int i = 0; i < 32; ++i) {
      int idx = i * 256 + tid;
      int m = idx >> 7, j = idx & 127;
      int rg = rowBase + m;
      int b = rg >> 14, t = rg & (Tlen - 1);
      int tt = t - 1 + tap;
      float v = (tt >= 0) ? x[((long)(b * Tlen + tt)) * 256 + icbase + j] : 0.0f;
      smA[m * 128 + j] = (__bf16)v;
    }
    async_wait0();
    __syncthreads();
#pragma unroll
    for (int s = 0; s < 2; ++s) {
      int u = wv + 8 * s;
      int m = (u & 3) * 16 + (lane & 15);
      int n = (u >> 2) * 16 + (lane & 15);
#pragma unroll
      for (int ks = 0; ks < 4; ++ks)
        acc[s] = wmma_bf16(frag_a(smA, 128, m, ks * 32, hi),
                           frag_b(smB, 128, n, ks * 32, hi), acc[s]);
    }
    __syncthreads();
  }
#pragma unroll
  for (int s = 0; s < 2; ++s) {
    int u = wv + 8 * s;
    int mi = u & 3;
    int n = (u >> 2) * 16 + (lane & 15);
    float bv = bias[n];
#pragma unroll
    for (int r = 0; r < 8; ++r) {
      int rg = rowBase + mi * 16 + r + 8 * hi;
      o[(long)rg * RC + n] = (__bf16)(acc[s][r] + bv);
    }
  }
}

// -------- residual block: gate conv 64->128 (K=2, dilated), tanh*sig, 1x1 64->64 --------
__global__ __launch_bounds__(256) void wn_res(const __bf16* __restrict__ o_in,
                                              __bf16* __restrict__ o_out,
                                              float* __restrict__ skip,
                                              const __bf16* __restrict__ w1t, // [128][128] Bt
                                              const float* __restrict__ b1,
                                              const __bf16* __restrict__ w2t, // [64][64] Bt
                                              const float* __restrict__ b2,
                                              int dil, int initSkip) {
  __shared__ __bf16 smA[64 * 128];   // [m][k]: k<64 tap(t-dil), k>=64 tap(t); reused for z
  __shared__ __bf16 smB[128 * 128];
  __shared__ __bf16 smW2[64 * 64];
  const int tid = threadIdx.x, lane = tid & 31, hi = lane >> 4, wv = tid >> 5;
  const int rowBase = blockIdx.x * 64;

  // weights: async (8 + 2 transfers/thread)
  for (int s = 0; s < 8; ++s) {
    int tIdx = s * 256 + tid;
    int n = tIdx >> 4, cc = tIdx & 15;
    async_ld16(w1t + n * 128 + cc * 8, smB + n * 128 + cc * 8);
  }
  for (int s = 0; s < 2; ++s) {
    int tIdx = s * 256 + tid;
    int n = tIdx >> 3, cc = tIdx & 7;
    async_ld16(w2t + n * 64 + cc * 8, smW2 + n * 64 + cc * 8);
  }
  // activations, current tap (k 64..127): async, rows of 128B -> 8 x b128
  for (int s = 0; s < 2; ++s) {
    int tIdx = s * 256 + tid;
    int m = tIdx >> 3, cc = tIdx & 7;
    async_ld16(o_in + (long)(rowBase + m) * RC + cc * 8, smA + m * 128 + 64 + cc * 8);
  }
  // dilated-history tap (k 0..63): async when whole tile is past the pad (block-uniform)
  if ((rowBase & (Tlen - 1)) >= dil) {
    for (int s = 0; s < 2; ++s) {
      int tIdx = s * 256 + tid;
      int m = tIdx >> 3, cc = tIdx & 7;
      async_ld16(o_in + (long)(rowBase + m - dil) * RC + cc * 8, smA + m * 128 + cc * 8);
    }
  } else {
    for (int i = 0; i < 16; ++i) {
      int idx = i * 256 + tid;
      int m = idx >> 6, k = idx & 63;
      int rg = rowBase + m;
      int t = rg & (Tlen - 1);
      smA[m * 128 + k] = (t >= dil) ? o_in[(long)(rg - dil) * RC + k] : (__bf16)0.0f;
    }
  }
  async_wait0();
  __syncthreads();

  const int mi = wv & 3;
  const int m = mi * 16 + (lane & 15);
  float zreg[2][8];
#pragma unroll
  for (int uu = 0; uu < 2; ++uu) {       // matched tanh/sigmoid tile pairs per wave
    int na = ((wv >> 2) + 2 * uu) * 16 + (lane & 15);
    int ng = 64 + na;
    v8f accA = {}, accG = {};
#pragma unroll
    for (int ks = 0; ks < 4; ++ks) {
      v16bf a = frag_a(smA, 128, m, ks * 32, hi);
      accA = wmma_bf16(a, frag_b(smB, 128, na, ks * 32, hi), accA);
      accG = wmma_bf16(a, frag_b(smB, 128, ng, ks * 32, hi), accG);
    }
    float biasA = b1[na], biasG = b1[ng];
#pragma unroll
    for (int r = 0; r < 8; ++r) {
      float av = accA[r] + biasA;
      float gv = accG[r] + biasG;
      zreg[uu][r] = tanhf(av) * (1.0f / (1.0f + __expf(-gv)));
    }
  }
  __syncthreads();                       // smA consumed -> reuse for z
  __bf16* smZ = smA;
#pragma unroll
  for (int uu = 0; uu < 2; ++uu) {
    int j = ((wv >> 2) + 2 * uu) * 16 + (lane & 15);
#pragma unroll
    for (int r = 0; r < 8; ++r)
      smZ[(mi * 16 + r + 8 * hi) * 64 + j] = (__bf16)zreg[uu][r];
  }
  __syncthreads();

#pragma unroll
  for (int uu = 0; uu < 2; ++uu) {       // 1x1: s = z @ W2t + b2
    int n = ((wv >> 2) + 2 * uu) * 16 + (lane & 15);
    v8f acc = {};
#pragma unroll
    for (int ks = 0; ks < 2; ++ks)
      acc = wmma_bf16(frag_a(smZ, 64, m, ks * 32, hi),
                      frag_b(smW2, 64, n, ks * 32, hi), acc);
    float bv = b2[n];
#pragma unroll
    for (int r = 0; r < 8; ++r) {
      int rg = rowBase + mi * 16 + r + 8 * hi;
      long off = (long)rg * RC + n;
      float sv = acc[r] + bv;
      o_out[off] = (__bf16)((float)o_in[off] + sv);
      if (initSkip) skip[off] = sv;
      else          skip[off] += sv;
    }
  }
}

// ------------- output: relu -> 1x1 64->256 -> relu -> 1x1 256->256 -------------
__global__ __launch_bounds__(256) void wn_out(const float* __restrict__ skip,
                                              const __bf16* __restrict__ wO1, // [256][64] Bt
                                              const float* __restrict__ b1,
                                              const __bf16* __restrict__ wO2, // [256][256] Bt
                                              const float* __restrict__ b2,
                                              float* __restrict__ out) {
  __shared__ __bf16 smR[32 * 64];
  __shared__ __bf16 smW[256 * 64];   // W1t, then W2 K-chunks
  __shared__ __bf16 smT[32 * 256];
  const int tid = threadIdx.x, lane = tid & 31, hi = lane >> 4, wv = tid >> 5;
  const int rowBase = blockIdx.x * 32;

  for (int s = 0; s < 8; ++s) {            // W1t async: 2048 x b128
    int tIdx = s * 256 + tid;
    int n = tIdx >> 3, cc = tIdx & 7;
    async_ld16(wO1 + n * 64 + cc * 8, smW + n * 64 + cc * 8);
  }
  for (int i = 0; i < 8; ++i) {            // relu(skip) fp32 -> bf16
    int idx = i * 256 + tid;
    smR[idx] = (__bf16)fmaxf(skip[(long)rowBase * 64 + idx], 0.0f);
  }
  async_wait0();
  __syncthreads();

#pragma unroll
  for (int s = 0; s < 4; ++s) {            // 32 tiles (2 mi x 16 ni), 4 per wave
    int u = wv + 8 * s;
    int mi = u & 1;
    int m = mi * 16 + (lane & 15);
    int n = (u >> 1) * 16 + (lane & 15);
    v8f acc = {};
#pragma unroll
    for (int ks = 0; ks < 2; ++ks)
      acc = wmma_bf16(frag_a(smR, 64, m, ks * 32, hi),
                      frag_b(smW, 64, n, ks * 32, hi), acc);
    float bv = b1[n];
#pragma unroll
    for (int r = 0; r < 8; ++r)
      smT[(mi * 16 + r + 8 * hi) * 256 + n] = (__bf16)fmaxf(acc[r] + bv, 0.0f);
  }
  __syncthreads();

  v8f acc2[4] = {};
  for (int c = 0; c < 4; ++c) {            // K = 256 in chunks of 64
    for (int s = 0; s < 8; ++s) {          // W2 chunk async: [256][64] bf16
      int tIdx = s * 256 + tid;
      int n = tIdx >> 3, cc = tIdx & 7;
      async_ld16(wO2 + n * 256 + c * 64 + cc * 8, smW + n * 64 + cc * 8);
    }
    async_wait0();
    __syncthreads();
#pragma unroll
    for (int s = 0; s < 4; ++s) {
      int u = wv + 8 * s;
      int m = (u & 1) * 16 + (lane & 15);
      int n = (u >> 1) * 16 + (lane & 15);
#pragma unroll
      for (int ks = 0; ks < 2; ++ks)
        acc2[s] = wmma_bf16(frag_a(smT, 256, m, c * 64 + ks * 32, hi),
                            frag_b(smW, 64, n, ks * 32, hi), acc2[s]);
    }
    __syncthreads();
  }
#pragma unroll
  for (int s = 0; s < 4; ++s) {
    int u = wv + 8 * s;
    int mi = u & 1;
    int n = (u >> 1) * 16 + (lane & 15);
    float bv = b2[n];
#pragma unroll
    for (int r = 0; r < 8; ++r) {
      int rg = rowBase + mi * 16 + r + 8 * hi;
      int b = rg >> 14, t = rg & (Tlen - 1);
      out[((long)b * OCH + n) * Tlen + t] = acc2[s][r] + bv;
    }
  }
}

extern "C" void kernel_launch(void* const* d_in, const int* in_sizes, int n_in,
                              void* d_out, int out_size, void* d_ws, size_t ws_size,
                              hipStream_t stream) {
  (void)in_sizes; (void)n_in; (void)out_size; (void)ws_size;
  const float* x     = (const float*)d_in[0];
  const float* il_w  = (const float*)d_in[1];
  const float* il_b  = (const float*)d_in[2];
  const float* rb_w1 = (const float*)d_in[3];
  const float* rb_b1 = (const float*)d_in[4];
  const float* rb_w2 = (const float*)d_in[5];
  const float* rb_b2 = (const float*)d_in[6];
  const float* ol_w1 = (const float*)d_in[7];
  const float* ol_b1 = (const float*)d_in[8];
  const float* ol_w2 = (const float*)d_in[9];
  const float* ol_b2 = (const float*)d_in[10];
  float* out = (float*)d_out;

  // workspace: o ping-pong (bf16) + skip (fp32) + bf16 weights  (~68 MB)
  __bf16* oA   = (__bf16*)d_ws;
  __bf16* oB   = oA + (size_t)R * RC;
  float*  skip = (float*)(oB + (size_t)R * RC);
  __bf16* wb   = (__bf16*)(skip + (size_t)R * RC);

  wn_cvt<<<W_TOT / 256, 256, 0, stream>>>(il_w, rb_w1, rb_w2, ol_w1, ol_w2, wb);
  wn_input<<<R / 64, 256, 0, stream>>>(x, wb + W_IL, il_b, oA);
  const __bf16* cur = oA;
  __bf16* nxt = oB;
  for (int i = 0; i < 20; ++i) {
    int dil = 1 << (i % 10);
    wn_res<<<R / 64, 256, 0, stream>>>(cur, nxt, skip,
        wb + W_G + (size_t)i * 128 * 128, rb_b1 + (size_t)i * 128,
        wb + W_S + (size_t)i * 64 * 64,   rb_b2 + (size_t)i * 64,
        dil, i == 0 ? 1 : 0);
    const __bf16* t = nxt; nxt = (__bf16*)cur; cur = t;   // ping-pong o
  }
  wn_out<<<R / 32, 256, 0, stream>>>(skip, wb + W_O1, ol_b1, wb + W_O2, ol_b2, out);
}
